// RAIN_1262720385108
// MI455X (gfx1250) — matrix operand order repaired and checked by reading, not compiled, not run on hardware
//
#include <hip/hip_runtime.h>
#include <math.h>

typedef __attribute__((ext_vector_type(16))) _Float16 v16h;
typedef __attribute__((ext_vector_type(8)))  float    v8f;

#define BB 16
#define TT 64
#define NN 64
#define SS 4
#define DD 128
#define HH 8
#define GG 384          // 3*D gates
#define BN 1024         // B*N
#define MTOT (TT*BN)    // 65536 rows for QKV GEMMs

__device__ __forceinline__ float sigmoidf_(float x) { return 1.0f / (1.0f + expf(-x)); }

// ---------------- CDNA5 async global->LDS copy (ASYNCcnt-tracked) ----------------
// CDNA5 ISA: GLOBAL_LOAD_ASYNC_TO_LDS_B128; LDS dest addr in VDST VGPR.
// Generic LDS pointer low 32 bits == LDS byte offset (ISA §10.2 aperture mapping).
__device__ __forceinline__ void async_copy_b128(unsigned int lds_addr, const void* gptr) {
  asm volatile("global_load_async_to_lds_b128 %0, %1, off"
               :: "v"(lds_addr), "v"((unsigned long long)(uintptr_t)gptr)
               : "memory");
}
__device__ __forceinline__ void wait_async0() {
  asm volatile("s_wait_asynccnt 0x0" ::: "memory");
}

// ---------------- WMMA fragment helpers (wave32, 16x16x32 f16 -> f32) ----------------
// A matrix 16x32 (MxK), row-major source with row stride lda (halves).
__device__ __forceinline__ v16h load_a_frag(const _Float16* __restrict__ base, int lda,
                                            int k0, int lane) {
  v16h a;
  const int m = lane & 15;
  const int khi = (lane >> 4) << 3;  // +8 for upper half-wave
  const _Float16* row = base + m * lda + k0;
#pragma unroll
  for (int v = 0; v < 8; ++v) {
    int kk = ((v >= 4) ? 16 : 0) + khi + ((v & 3) << 1);
    a[2 * v]     = row[kk];
    a[2 * v + 1] = row[kk + 1];
  }
  return a;
}

// B matrix 32x16 (KxN) from a TRANSPOSED weight Wt[N][K] (row stride ldb halves).
// lane L: N=L&15; halves h=0..7 hold K = (L>=16?8:0)+h, h=8..15 hold K = 16+(L>=16?8:0)+(h-8).
__device__ __forceinline__ v16h load_bt_frag(const _Float16* __restrict__ wt, int ldb,
                                             int k0, int n0, int lane) {
  v16h b;
  const _Float16* col = wt + (size_t)(n0 + (lane & 15)) * ldb + k0 + ((lane >> 4) << 3);
#pragma unroll
  for (int h = 0; h < 16; ++h) b[h] = col[(h < 8) ? h : h + 8];
  return b;
}

__device__ __forceinline__ v8f wmma16(v16h a, v16h b, v8f c) {
  return __builtin_amdgcn_wmma_f32_16x16x32_f16(false, a, false, b, (short)0, c, false, false);
}

// ---------------- K0: weight convert + transpose to f16 ----------------
__global__ void convert_kernel(const float* __restrict__ W_ih, const float* __restrict__ W_hh,
                               const float* __restrict__ W_kc, const float* __restrict__ W_qc,
                               const float* __restrict__ W_vc,
                               _Float16* __restrict__ wih_t, _Float16* __restrict__ whh_t,
                               _Float16* __restrict__ wq_t, _Float16* __restrict__ wk_t,
                               _Float16* __restrict__ wv_t) {
  int idx = blockIdx.x * 256 + threadIdx.x;
  if (idx < DD * GG) {          // [128][384] -> [384][128]
    int k = idx / GG, n = idx % GG;
    wih_t[n * DD + k] = (_Float16)W_ih[idx];
    whh_t[n * DD + k] = (_Float16)W_hh[idx];
  }
  if (idx < DD * DD) {          // [128][128] -> transposed
    int k = idx / DD, n = idx % DD;
    wq_t[n * DD + k] = (_Float16)W_qc[idx];
    wk_t[n * DD + k] = (_Float16)W_kc[idx];
    wv_t[n * DD + k] = (_Float16)W_vc[idx];
  }
}

// ---------------- K1: state encoder (S=4 -> D=128) for frames 0..T, + h0 ----------------
__global__ void encode_kernel(const float* __restrict__ x,
                              const float* __restrict__ W_se, const float* __restrict__ b_se,
                              const float* __restrict__ W_init, const float* __restrict__ b_init,
                              _Float16* __restrict__ xe,   // [T][BN][D] f16
                              float* __restrict__ xd,      // [BN][D]
                              float* __restrict__ h0) {    // [BN][D]
  size_t idx = (size_t)blockIdx.x * 256 + threadIdx.x;
  const size_t total = (size_t)(TT + 1) * BN * DD;
  if (idx >= total) return;
  int dd = (int)(idx & (DD - 1));
  size_t r = idx >> 7;
  int bn = (int)(r % BN);
  int t = (int)(r / BN);
  int b = bn >> 6, n = bn & 63;
  const float* xp = x + (((size_t)b * (TT + 1) + t) * NN + n) * SS;
  float s = b_se[dd];
#pragma unroll
  for (int q = 0; q < SS; ++q) s += xp[q] * W_se[q * DD + dd];
  if (t < TT) xe[((size_t)t * BN + bn) * DD + dd] = (_Float16)s;
  else        xd[(size_t)bn * DD + dd] = s;
  if (t == 0) {
    float hi = b_init[dd];
#pragma unroll
    for (int q = 0; q < SS; ++q) hi += xp[q] * W_init[q * DD + dd];
    h0[(size_t)bn * DD + dd] = hi;
  }
}

// ---------------- K2: GRU recurrence, WMMA, weights resident in LDS ----------------
// 64 blocks x 256 threads; block owns rows [16*blk, 16*blk+16); wave w owns 16 gate cols.
__global__ void __launch_bounds__(256) gru_kernel(
    const _Float16* __restrict__ xe,     // [T][BN][D]
    const _Float16* __restrict__ wih_t,  // [384][128] f16 (transposed)
    const _Float16* __restrict__ whh_t,  // [384][128]
    const float* __restrict__ b_ih, const float* __restrict__ b_hh,
    const float* __restrict__ h0,
    _Float16* __restrict__ out,          // [T][BN][D] f16
    float* __restrict__ hT) {            // [BN][D]
  extern __shared__ _Float16 smem[];
  _Float16* s_wih = smem;                       // 384*128
  _Float16* s_whh = s_wih + GG * DD;            // 384*128
  _Float16* s_x   = s_whh + GG * DD;            // 16*128
  _Float16* s_h   = s_x + 16 * DD;              // 16*128

  const int tid = threadIdx.x;
  const int lane = tid & 31;
  const int wave = tid >> 5;
  const int m0 = blockIdx.x * 16;

  // async-stage both weight panels (192KB) into LDS, ASYNCcnt-tracked
  {
    const unsigned int swih = (unsigned int)(uintptr_t)s_wih;
    const unsigned int swhh = (unsigned int)(uintptr_t)s_whh;
    const char* gih = (const char*)wih_t;
    const char* ghh = (const char*)whh_t;
    for (int i = tid * 16; i < GG * DD * 2; i += 256 * 16) {
      async_copy_b128(swih + i, gih + i);
      async_copy_b128(swhh + i, ghh + i);
    }
  }
  // h init (f32 -> f16 conversion) overlaps with the async weight DMA
  for (int i = tid; i < 16 * DD; i += 256) {
    int rr = i >> 7, cc = i & (DD - 1);
    s_h[i] = (_Float16)h0[(size_t)(m0 + rr) * DD + cc];
  }

  const int ncol = wave * 16;           // this wave's 16-column slice of each gate section
  const int col  = ncol + (lane & 15);  // per-lane column (matches D-frag N)
  const float bir = b_ih[col],         bhr = b_hh[col];
  const float biz = b_ih[col + DD],    bhz = b_hh[col + DD];
  const float bin_ = b_ih[col + 2*DD], bhn = b_hh[col + 2*DD];

  const unsigned int sx_base = (unsigned int)(uintptr_t)s_x;

  for (int t = 0; t < TT; ++t) {
    // async-stage x_t tile (16x128 f16 = 4KB, one b128 per thread)
    const char* xtg = (const char*)(xe + ((size_t)t * BN + m0) * DD);
    async_copy_b128(sx_base + tid * 16, xtg + tid * 16);
    if (t + 1 < TT)  // prefetch next step's tile into cache hierarchy
      __builtin_prefetch((const char*)(xe + ((size_t)(t + 1) * BN + m0) * DD) + tid * 16, 0, 0);
    wait_async0();
    __syncthreads();   // x staged; h from previous step visible

    v8f gir = {}, giz = {}, gin = {}, ghr = {}, ghz = {}, ghn = {};
#pragma unroll
    for (int kk = 0; kk < DD; kk += 32) {
      v16h ax = load_a_frag(s_x, DD, kk, lane);
      v16h ah = load_a_frag(s_h, DD, kk, lane);
      gir = wmma16(ax, load_bt_frag(s_wih, DD, kk, ncol,          lane), gir);
      giz = wmma16(ax, load_bt_frag(s_wih, DD, kk, ncol + DD,     lane), giz);
      gin = wmma16(ax, load_bt_frag(s_wih, DD, kk, ncol + 2 * DD, lane), gin);
      ghr = wmma16(ah, load_bt_frag(s_whh, DD, kk, ncol,          lane), ghr);
      ghz = wmma16(ah, load_bt_frag(s_whh, DD, kk, ncol + DD,     lane), ghz);
      ghn = wmma16(ah, load_bt_frag(s_whh, DD, kk, ncol + 2 * DD, lane), ghn);
    }
    __syncthreads();   // everyone done reading s_x / s_h

#pragma unroll
    for (int r = 0; r < 8; ++r) {
      int M = r + ((lane >> 4) << 3);            // D-frag row mapping
      float rg = sigmoidf_(gir[r] + bir + ghr[r] + bhr);
      float zg = sigmoidf_(giz[r] + biz + ghz[r] + bhz);
      float ng = tanhf(gin[r] + bin_ + rg * (ghn[r] + bhn));
      float ho = (float)s_h[M * DD + col];       // own (M,col) only -> no cross-wave hazard
      float hn = (1.0f - zg) * ng + zg * ho;
      s_h[M * DD + col] = (_Float16)hn;
      out[((size_t)t * BN + m0 + M) * DD + col] = (_Float16)hn;
    }
  }
  __syncthreads();
  for (int i = tid; i < 16 * DD; i += 256) {
    int rr = i >> 7, cc = i & (DD - 1);
    hT[(size_t)(m0 + rr) * DD + cc] = (float)s_h[i];
  }
}

// ---------------- K3: fused Q/K/V projection GEMM (WMMA, async-staged LDS) ----------------
__global__ void __launch_bounds__(256) qkv_kernel(
    const _Float16* __restrict__ A,      // [M=65536][128] f16 (GRU outputs)
    const _Float16* __restrict__ wq_t, const _Float16* __restrict__ wk_t,
    const _Float16* __restrict__ wv_t,
    _Float16* __restrict__ q, _Float16* __restrict__ k, _Float16* __restrict__ v) {
  extern __shared__ _Float16 smem[];
  _Float16* s_a = smem;                 // 64*128
  _Float16* s_w = s_a + 64 * DD;        // 3*128*128, order q,k,v
  const int tid = threadIdx.x, lane = tid & 31, wave = tid >> 5;
  const size_t m0 = (size_t)blockIdx.x * 64;

  {
    const unsigned int sw = (unsigned int)(uintptr_t)s_w;
    const unsigned int sa = (unsigned int)(uintptr_t)s_a;
    for (int i = tid * 16; i < DD * DD * 2; i += 256 * 16) {
      async_copy_b128(sw + i,                 (const char*)wq_t + i);
      async_copy_b128(sw + DD * DD * 2 + i,   (const char*)wk_t + i);
      async_copy_b128(sw + 2 * DD * DD * 2 + i, (const char*)wv_t + i);
    }
    const char* ag = (const char*)(A + m0 * DD);
    for (int i = tid * 16; i < 64 * DD * 2; i += 256 * 16)
      async_copy_b128(sa + i, ag + i);
    wait_async0();
  }
  __syncthreads();

  for (int idx = wave; idx < 96; idx += 8) {   // 3 outputs x 4 Mtiles x 8 Ntiles
    int which = idx >> 5;            // wave-uniform
    int rem = idx & 31;
    int mt = rem >> 3, nt = rem & 7;
    v8f acc = {};
#pragma unroll
    for (int kk = 0; kk < DD; kk += 32) {
      v16h a = load_a_frag(s_a + mt * 16 * DD, DD, kk, lane);
      v16h b = load_bt_frag(s_w + which * DD * DD, DD, kk, nt * 16, lane);
      acc = wmma16(a, b, acc);
    }
    _Float16* dst = (which == 0) ? q : (which == 1) ? k : v;
#pragma unroll
    for (int r = 0; r < 8; ++r) {
      int M = mt * 16 + r + ((lane >> 4) << 3);
      int N = nt * 16 + (lane & 15);
      dst[(m0 + M) * DD + N] = (_Float16)acc[r];
    }
  }
}

// ---------------- K4: per-pair temporal attention, fused softmax ----------------
// block = (b,h,i); thread = j. score[b,h,i,j,t] = q[t,b,j,h,:]·k[t,b,i,h,:]/sqrt(D)
__global__ void __launch_bounds__(64) attn_kernel(
    const _Float16* __restrict__ q, const _Float16* __restrict__ k,
    const _Float16* __restrict__ v, float* __restrict__ ctx) {  // ctx [B][N][N][D]
  const int j = threadIdx.x;
  const int bhi = blockIdx.x;
  const int i = bhi & 63, h = (bhi >> 6) & 7, b = bhi >> 9;
  const float scale = 0.088388347648318447f;  // 1/sqrt(128)
  float sc[TT];
#pragma unroll
  for (int t = 0; t < TT; ++t) {
    const _Float16* qp = q + ((size_t)t * BN + b * NN + j) * DD + h * 16;
    const _Float16* kp = k + ((size_t)t * BN + b * NN + i) * DD + h * 16;
    float s = 0.f;
#pragma unroll
    for (int d = 0; d < 16; ++d) s += (float)qp[d] * (float)kp[d];
    sc[t] = s * scale;
  }
  float mx = sc[0];
#pragma unroll
  for (int t = 1; t < TT; ++t) mx = fmaxf(mx, sc[t]);
  float sum = 0.f;
#pragma unroll
  for (int t = 0; t < TT; ++t) { sc[t] = expf(sc[t] - mx); sum += sc[t]; }
  float inv = 1.0f / sum;
  float acc[16];
#pragma unroll
  for (int d = 0; d < 16; ++d) acc[d] = 0.f;
#pragma unroll
  for (int t = 0; t < TT; ++t) {
    float a = sc[t] * inv;
    const _Float16* vp = v + ((size_t)t * BN + b * NN + i) * DD + h * 16;
#pragma unroll
    for (int d = 0; d < 16; ++d) acc[d] += a * (float)vp[d];
  }
  float* cp = ctx + ((size_t)(b * NN + i) * NN + j) * DD + h * 16;
#pragma unroll
  for (int d = 0; d < 16; ++d) cp[d] = acc[d];
}

// ---------------- K5: GAT edge weights ----------------
__global__ void gat_kernel(const float* __restrict__ ctx, const float* __restrict__ W_att,
                           const float* __restrict__ b_att, float* __restrict__ weight) {
  int idx = blockIdx.x * 256 + threadIdx.x;      // B*N*N = 65536
  int j = idx & 63, i = (idx >> 6) & 63, b = idx >> 12;
  const float* c1 = ctx + (size_t)idx * DD;                                   // ctx[b,i,j,:]
  const float* c2 = ctx + ((size_t)(b * NN + j) * NN + i) * DD;               // ctx[b,j,i,:]
  float e = b_att[0];
  for (int c = 0; c < DD; ++c) e += c1[c] * W_att[c] + c2[c] * W_att[DD + c];
  weight[idx] = (i == j) ? 0.f : tanhf(e + 0.5f);
}

// ---------------- K6: decode GRU step + value head (per row) ----------------
__global__ void __launch_bounds__(128) decode_kernel(
    const float* __restrict__ xd, const float* __restrict__ hT,
    const float* __restrict__ W_ih, const float* __restrict__ W_hh,
    const float* __restrict__ b_ih, const float* __restrict__ b_hh,
    const float* __restrict__ W_val, const float* __restrict__ b_val,
    float* __restrict__ h2, float* __restrict__ vv) {
  __shared__ float s_x[DD], s_h[DD], s_h2[DD];
  const int bn = blockIdx.x, dd = threadIdx.x;
  s_x[dd] = xd[(size_t)bn * DD + dd];
  s_h[dd] = hT[(size_t)bn * DD + dd];
  __syncthreads();
  float gir = b_ih[dd], giz = b_ih[dd + DD], gin = b_ih[dd + 2 * DD];
  float ghr = b_hh[dd], ghz = b_hh[dd + DD], ghn = b_hh[dd + 2 * DD];
  for (int c = 0; c < DD; ++c) {
    float xv = s_x[c], hv = s_h[c];
    gir += xv * W_ih[c * GG + dd];
    giz += xv * W_ih[c * GG + dd + DD];
    gin += xv * W_ih[c * GG + dd + 2 * DD];
    ghr += hv * W_hh[c * GG + dd];
    ghz += hv * W_hh[c * GG + dd + DD];
    ghn += hv * W_hh[c * GG + dd + 2 * DD];
  }
  float r = sigmoidf_(gir + ghr), z = sigmoidf_(giz + ghz);
  float n = tanhf(gin + r * ghn);
  float hn = (1.0f - z) * n + z * s_h[dd];
  s_h2[dd] = hn;
  h2[(size_t)bn * DD + dd] = hn;
  __syncthreads();
  float o = b_val[dd];
  for (int c = 0; c < DD; ++c) o += s_h2[c] * W_val[c * DD + dd];
  vv[(size_t)bn * DD + dd] = o;
}

// ---------------- K7: aggregate + dec + mu/sig heads (per (b,i) row) ----------------
__global__ void __launch_bounds__(128) head_kernel(
    const float* __restrict__ weight, const float* __restrict__ vv,
    const float* __restrict__ W_dec, const float* __restrict__ b_dec,
    const float* __restrict__ W_mu, const float* __restrict__ b_mu,
    const float* __restrict__ W_sig, const float* __restrict__ b_sig,
    float* __restrict__ outp) {  // [mu(4096) | sig(4096)]
  __shared__ float s_w[NN], s_v[DD], s_p[DD], s_d[DD];
  const int bi = blockIdx.x;       // b*N+i
  const int b = bi >> 6;
  const int dd = threadIdx.x;
  if (dd < NN) s_w[dd] = weight[(size_t)bi * NN + dd];
  s_v[dd] = vv[(size_t)bi * DD + dd];
  __syncthreads();
  float p = 0.f;
  for (int j = 0; j < NN; ++j) p += s_w[j] * vv[((size_t)b * NN + j) * DD + dd];
  s_p[dd] = p * (1.0f / (float)NN);
  __syncthreads();
  float dv = b_dec[dd];
  for (int c = 0; c < DD; ++c) dv += s_v[c] * W_dec[c * DD + dd] + s_p[c] * W_dec[(c + DD) * DD + dd];
  s_d[dd] = dv;
  __syncthreads();
  if (dd < SS) {
    float m = b_mu[dd];
    for (int c = 0; c < DD; ++c) m += s_d[c] * W_mu[c * SS + dd];
    outp[(size_t)bi * SS + dd] = m;
  } else if (dd < 2 * SS) {
    int s4 = dd - SS;
    float g = b_sig[s4];
    for (int c = 0; c < DD; ++c) g += s_d[c] * W_sig[c * SS + s4];
    outp[(size_t)BB * NN * SS + (size_t)bi * SS + s4] = sigmoidf_(g) + 1e-6f;
  }
}

// ---------------- launch ----------------
extern "C" void kernel_launch(void* const* d_in, const int* in_sizes, int n_in,
                              void* d_out, int out_size, void* d_ws, size_t ws_size,
                              hipStream_t stream) {
  const float* x      = (const float*)d_in[0];
  const float* W_se   = (const float*)d_in[1];
  const float* b_se   = (const float*)d_in[2];
  const float* W_init = (const float*)d_in[3];
  const float* b_init = (const float*)d_in[4];
  const float* W_ih   = (const float*)d_in[5];
  const float* W_hh   = (const float*)d_in[6];
  const float* b_ih   = (const float*)d_in[7];
  const float* b_hh   = (const float*)d_in[8];
  const float* W_kc   = (const float*)d_in[9];
  const float* W_qc   = (const float*)d_in[10];
  const float* W_vc   = (const float*)d_in[11];
  const float* W_att  = (const float*)d_in[12];
  const float* b_att  = (const float*)d_in[13];
  const float* W_val  = (const float*)d_in[14];
  const float* b_val  = (const float*)d_in[15];
  const float* W_dec  = (const float*)d_in[16];
  const float* b_dec  = (const float*)d_in[17];
  const float* W_mu   = (const float*)d_in[18];
  const float* b_mu   = (const float*)d_in[19];
  const float* W_sig  = (const float*)d_in[20];
  const float* b_sig  = (const float*)d_in[21];

  // ---- workspace partition (halves first, then floats; all 16B-aligned) ----
  _Float16* hbase = (_Float16*)d_ws;
  size_t ho = 0;
  _Float16* wih_t = hbase + ho; ho += (size_t)GG * DD;       // 49152
  _Float16* whh_t = hbase + ho; ho += (size_t)GG * DD;
  _Float16* wq_t  = hbase + ho; ho += (size_t)DD * DD;
  _Float16* wk_t  = hbase + ho; ho += (size_t)DD * DD;
  _Float16* wv_t  = hbase + ho; ho += (size_t)DD * DD;
  _Float16* xe    = hbase + ho; ho += (size_t)TT * BN * DD;  // 8.39M halves
  _Float16* gout  = hbase + ho; ho += (size_t)TT * BN * DD;
  _Float16* qb    = hbase + ho; ho += (size_t)TT * BN * DD;
  _Float16* kb    = hbase + ho; ho += (size_t)TT * BN * DD;
  _Float16* vb    = hbase + ho; ho += (size_t)TT * BN * DD;
  size_t hbytes = (ho * sizeof(_Float16) + 255) & ~(size_t)255;
  float* fbase = (float*)((char*)d_ws + hbytes);
  size_t fo = 0;
  float* xd  = fbase + fo; fo += (size_t)BN * DD;
  float* h0  = fbase + fo; fo += (size_t)BN * DD;
  float* hT  = fbase + fo; fo += (size_t)BN * DD;
  float* h2  = fbase + fo; fo += (size_t)BN * DD;
  float* vv  = fbase + fo; fo += (size_t)BN * DD;
  float* ctx = fbase + fo; fo += (size_t)BB * NN * NN * DD;  // 8.39M floats
  float* wgt = fbase + fo; fo += (size_t)BB * NN * NN;

  float* outp = (float*)d_out;

  // K0: weight convert
  convert_kernel<<<(DD * GG + 255) / 256, 256, 0, stream>>>(
      W_ih, W_hh, W_kc, W_qc, W_vc, wih_t, whh_t, wq_t, wk_t, wv_t);

  // K1: encoder, (T+1)*BN*D threads
  {
    size_t total = (size_t)(TT + 1) * BN * DD;
    encode_kernel<<<(unsigned)((total + 255) / 256), 256, 0, stream>>>(
        x, W_se, b_se, W_init, b_init, xe, xd, h0);
  }

  // K2: GRU, 64 blocks, 200KB dynamic LDS (weights + x/h tiles)
  {
    size_t lds = ((size_t)2 * GG * DD + 2 * 16 * DD) * sizeof(_Float16);  // 204800 B
    gru_kernel<<<BN / 16, 256, lds, stream>>>(xe, wih_t, whh_t, b_ih, b_hh, h0, gout, hT);
  }

  // K3: Q/K/V projections, 1024 blocks of 64 rows, 112KB LDS
  {
    size_t lds = ((size_t)64 * DD + 3 * DD * DD) * sizeof(_Float16);  // 114688 B
    qkv_kernel<<<MTOT / 64, 256, lds, stream>>>(gout, wq_t, wk_t, wv_t, qb, kb, vb);
  }

  // K4: attention, one block per (b,h,i)
  attn_kernel<<<BB * HH * NN, 64, 0, stream>>>(qb, kb, vb, ctx);

  // K5: GAT edge weights
  gat_kernel<<<(BB * NN * NN) / 256, 256, 0, stream>>>(ctx, W_att, b_att, wgt);

  // K6: decode GRU step + value head
  decode_kernel<<<BN, DD, 0, stream>>>(xd, hT, W_ih, W_hh, b_ih, b_hh, W_val, b_val, h2, vv);

  // K7: aggregate + heads
  head_kernel<<<BB * NN, DD, 0, stream>>>(wgt, vv, W_dec, b_dec, W_mu, b_mu, W_sig, b_sig, outp);
}
